// AtomAttentionEncoder_25305947308317
// MI455X (gfx1250) — matrix operand order repaired and checked by reading, not compiled, not run on hardware
//
#include <hip/hip_runtime.h>
#include <hip/hip_bf16.h>

// ---------------------------------------------------------------------------
// Types
// ---------------------------------------------------------------------------
typedef _Float16 h16;
typedef __attribute__((ext_vector_type(16))) _Float16 v16h;
typedef __attribute__((ext_vector_type(4)))  _Float16 v4h;
typedef __attribute__((ext_vector_type(8)))  float    v8f;

#define NTOK   16384
#define MROWS  32768     // B * N
#define CA_    128
#define CS_    128
#define CZ_    16
#define H_     4
#define DH_    32
#define NQ_    32
#define NK_    128
#define NB_    512

__device__ __forceinline__ float sigmoidf_(float x) {
    return 1.0f / (1.0f + __expf(-x));
}

__device__ __forceinline__ v8f wmma_f16(v16h a, v16h b, v8f c) {
    return __builtin_amdgcn_wmma_f32_16x16x32_f16(
        /*neg_a=*/false, a, /*neg_b=*/false, b,
        /*c_mod=*/(short)0, c, /*reuse_a=*/false, /*reuse_b=*/false);
}

// A-fragment (16x32, f16), row-major source, contiguous K order:
//   elem i -> K = kbase + 16*half + i
// -> two b128 loads per lane.
__device__ __forceinline__ v16h load_a_frag(const h16* A, long row, int ld,
                                            int kbase, int lane) {
    int half = (lane >> 4) & 1;
    const h16* p = A + row * (long)ld + kbase + 16 * half;
    union { uint4 u[2]; v16h v; } r;
    r.u[0] = *(const uint4*)(p);
    r.u[1] = *(const uint4*)(p + 8);
    return r.v;
}

// B-fragment (32x16, f16) from TRANSPOSED weight Wt[N][K] (ld = K):
//   lane col n = col + lane&15; elem i -> K = kbase + 16*half + i (contiguous)
// -> two b128 loads per lane.
__device__ __forceinline__ v16h load_bt_frag(const h16* Wt, int kbase, int col,
                                             int K, int lane) {
    int half = (lane >> 4) & 1;
    int n = col + (lane & 15);
    const h16* p = Wt + (long)n * K + kbase + 16 * half;
    union { uint4 u[2]; v16h v; } r;
    r.u[0] = *(const uint4*)(p);
    r.u[1] = *(const uint4*)(p + 8);
    return r.v;
}

// B-fragment for attention K-matrix: B[K=dh][n=key]; dh contiguous per token.
__device__ __forceinline__ v16h load_kb_frag(const h16* Km, long tok,
                                             int chbase, int half) {
    const h16* p = Km + tok * CA_ + chbase + 16 * half;
    union { uint4 u[2]; v16h v; } r;
    r.u[0] = *(const uint4*)(p);
    r.u[1] = *(const uint4*)(p + 8);
    return r.v;
}

// B-fragment for attention V from TRANSPOSED vt[ch][MROWS]:
//   lane n = dh channel; K elements = contiguous tokens (vector fast path,
//   clamped scalar fallback only at sequence edges).
__device__ __forceinline__ v16h load_vt_frag(const h16* Vt, long chbase,
                                             int k0, int ntok) {
    if (k0 >= 0 && k0 + 15 < ntok) {
        const h16* p = Vt + chbase + k0;
        union { uint4 u[2]; v16h v; } r;
        r.u[0] = *(const uint4*)(p);
        r.u[1] = *(const uint4*)(p + 8);
        return r.v;
    }
    union { h16 e[16]; v16h v; } r;
#pragma unroll
    for (int i = 0; i < 16; ++i) {
        int gk = k0 + i;
        gk = gk < 0 ? 0 : (gk >= ntok ? ntok - 1 : gk);
        r.e[i] = Vt[chbase + gk];
    }
    return r.v;
}

// ---------------------------------------------------------------------------
// Kernel 1: f32 -> f16 with transpose: src[K][N] -> dst[N][K]
// ---------------------------------------------------------------------------
__global__ void k_f32_to_f16_t(const float* __restrict__ src, h16* __restrict__ dst,
                               int K, int N) {
    int i = blockIdx.x * blockDim.x + threadIdx.x;
    if (i < K * N) {
        int k = i / N, n = i - k * N;
        dst[(long)n * K + k] = (h16)src[i];
    }
}

// ---------------------------------------------------------------------------
// Kernel 2: per-row LayerNorms of a and s (one wave per 128-wide row)
// ---------------------------------------------------------------------------
__global__ void k_rowln(const float* __restrict__ a, const float* __restrict__ s,
                        const float* __restrict__ gapb, const float* __restrict__ gct,
                        float* __restrict__ a_ln, h16* __restrict__ sn_apb,
                        h16* __restrict__ sn_ct, h16* __restrict__ s_h) {
    int row  = blockIdx.x * (blockDim.x >> 5) + (threadIdx.x >> 5);
    int lane = threadIdx.x & 31;
    const float4 av = ((const float4*)(a + (long)row * CA_))[lane];
    const float4 sv = ((const float4*)(s + (long)row * CS_))[lane];

    float sa  = av.x + av.y + av.z + av.w;
    float sa2 = av.x*av.x + av.y*av.y + av.z*av.z + av.w*av.w;
    float ss  = sv.x + sv.y + sv.z + sv.w;
    float ss2 = sv.x*sv.x + sv.y*sv.y + sv.z*sv.z + sv.w*sv.w;
#pragma unroll
    for (int m = 16; m >= 1; m >>= 1) {
        sa  += __shfl_xor(sa,  m, 32);
        sa2 += __shfl_xor(sa2, m, 32);
        ss  += __shfl_xor(ss,  m, 32);
        ss2 += __shfl_xor(ss2, m, 32);
    }
    float ma = sa * (1.0f/128.0f), ms = ss * (1.0f/128.0f);
    float ra = rsqrtf(sa2 * (1.0f/128.0f) - ma*ma + 1e-5f);
    float rs = rsqrtf(ss2 * (1.0f/128.0f) - ms*ms + 1e-5f);

    int c = lane * 4;
    float4 an;
    an.x = (av.x-ma)*ra; an.y = (av.y-ma)*ra; an.z = (av.z-ma)*ra; an.w = (av.w-ma)*ra;
    ((float4*)(a_ln + (long)row * CA_))[lane] = an;

    float snv[4]  = { (sv.x-ms)*rs, (sv.y-ms)*rs, (sv.z-ms)*rs, (sv.w-ms)*rs };
    float sraw[4] = { sv.x, sv.y, sv.z, sv.w };
    v4h t0, t1, t2;
#pragma unroll
    for (int j = 0; j < 4; ++j) {
        t0[j] = (h16)(snv[j] * gapb[c + j]);
        t1[j] = (h16)(snv[j] * gct[c + j]);
        t2[j] = (h16)sraw[j];
    }
    *(v4h*)(sn_apb + (long)row * CS_ + c) = t0;
    *(v4h*)(sn_ct  + (long)row * CS_ + c) = t1;
    *(v4h*)(s_h    + (long)row * CS_ + c) = t2;
}

// ---------------------------------------------------------------------------
// Kernel 3: adaLN dual-GEMM, 4 row tiles per wave (B-frag reuse x4):
//   out = sigmoid(sn@W1 + bg) * a_ln + sn@W2  (f16)
// ---------------------------------------------------------------------------
__global__ void k_adaln_gemm(const h16* __restrict__ A, const h16* __restrict__ W1t,
                             const h16* __restrict__ W2t, const float* __restrict__ bg,
                             const float* __restrict__ a_ln, h16* __restrict__ out) {
    int wid  = blockIdx.x * (blockDim.x >> 5) + (threadIdx.x >> 5);
    int lane = threadIdx.x & 31;
    int half = lane >> 4, l15 = lane & 15;
    int tg = wid >> 3, tn = wid & 7;
    long row0 = (long)tg * 64 + l15;
    v8f c1[4], c2[4];
#pragma unroll
    for (int j = 0; j < 4; ++j) { c1[j] = (v8f)(0.f); c2[j] = (v8f)(0.f); }
#pragma unroll
    for (int k = 0; k < CS_; k += 32) {
        v16h b1 = load_bt_frag(W1t, k, tn * 16, CS_, lane);
        v16h b2 = load_bt_frag(W2t, k, tn * 16, CS_, lane);
#pragma unroll
        for (int j = 0; j < 4; ++j) {
            v16h af = load_a_frag(A, row0 + j * 16, CS_, k, lane);
            c1[j] = wmma_f16(af, b1, c1[j]);
            c2[j] = wmma_f16(af, b2, c2[j]);
        }
    }
    int n = tn * 16 + l15;
#pragma unroll
    for (int j = 0; j < 4; ++j)
#pragma unroll
        for (int r = 0; r < 8; ++r) {
            long orow = (long)tg * 64 + j * 16 + r + 8 * half;
            float v = sigmoidf_(c1[j][r] + bg[n]) * a_ln[orow * CA_ + n] + c2[j][r];
            out[orow * CA_ + n] = (h16)v;
        }
}

// ---------------------------------------------------------------------------
// Kernel 4: quad-GEMM from a1, 2 row tiles per wave:
//   q (scaled,+bq) f16, k f16, v -> TRANSPOSED vt[ch][tok] f16, gate g f32
// ---------------------------------------------------------------------------
__global__ void k_qkvg_gemm(const h16* __restrict__ A,
                            const h16* __restrict__ Wqt, const h16* __restrict__ Wkt,
                            const h16* __restrict__ Wvt, const h16* __restrict__ Wgt,
                            const float* __restrict__ bq,
                            h16* __restrict__ qo, h16* __restrict__ ko,
                            h16* __restrict__ vt, float* __restrict__ go) {
    int wid  = blockIdx.x * (blockDim.x >> 5) + (threadIdx.x >> 5);
    int lane = threadIdx.x & 31;
    int half = lane >> 4, l15 = lane & 15;
    int tg = wid >> 3, tn = wid & 7;
    long row0 = (long)tg * 32 + l15;
    v8f cq[2], ck[2], cv[2], cg[2];
#pragma unroll
    for (int j = 0; j < 2; ++j) {
        cq[j] = (v8f)(0.f); ck[j] = (v8f)(0.f);
        cv[j] = (v8f)(0.f); cg[j] = (v8f)(0.f);
    }
#pragma unroll
    for (int k = 0; k < CA_; k += 32) {
        v16h bqf = load_bt_frag(Wqt, k, tn * 16, CA_, lane);
        v16h bkf = load_bt_frag(Wkt, k, tn * 16, CA_, lane);
        v16h bvf = load_bt_frag(Wvt, k, tn * 16, CA_, lane);
        v16h bgf = load_bt_frag(Wgt, k, tn * 16, CA_, lane);
#pragma unroll
        for (int j = 0; j < 2; ++j) {
            v16h af = load_a_frag(A, row0 + j * 16, CA_, k, lane);
            cq[j] = wmma_f16(af, bqf, cq[j]);
            ck[j] = wmma_f16(af, bkf, ck[j]);
            cv[j] = wmma_f16(af, bvf, cv[j]);
            cg[j] = wmma_f16(af, bgf, cg[j]);
        }
    }
    int n = tn * 16 + l15;
    const float qscale = 0.17677669529663687f;  // 1/sqrt(32)
#pragma unroll
    for (int j = 0; j < 2; ++j) {
        long rbase = (long)tg * 32 + j * 16 + 8 * half;  // 8 consecutive rows
#pragma unroll
        for (int r = 0; r < 8; ++r) {
            long orow = rbase + r;
            qo[orow * CA_ + n] = (h16)((cq[j][r] + bq[n]) * qscale);
            ko[orow * CA_ + n] = (h16)ck[j][r];
            go[orow * CA_ + n] = sigmoidf_(cg[j][r]);
        }
        // v transposed: lane's 8 rows are contiguous tokens -> two b64 stores
        v4h vlo, vhi;
#pragma unroll
        for (int r = 0; r < 4; ++r) { vlo[r] = (h16)cv[j][r]; vhi[r] = (h16)cv[j][r + 4]; }
        long vb = (long)n * MROWS + rbase;
        *(v4h*)(vt + vb)     = vlo;
        *(v4h*)(vt + vb + 4) = vhi;
    }
}

// ---------------------------------------------------------------------------
// Kernel 5: pair bias, stored TRANSPOSED: bias_t[b,h,nb,k,q]
// Streams the 268 MB z tensor once (HBM-bound stage).
// ---------------------------------------------------------------------------
__global__ void k_bias(const float* __restrict__ z, const float* __restrict__ gz,
                       const float* __restrict__ Wz, float* __restrict__ bias_t) {
    int idx = blockIdx.x * blockDim.x + threadIdx.x;  // (b,nb,q,k) flat
    const float4* zp = (const float4*)(z + (long)idx * CZ_);
    float4 z0 = zp[0], z1 = zp[1], z2 = zp[2], z3 = zp[3];
    float x[16] = { z0.x, z0.y, z0.z, z0.w, z1.x, z1.y, z1.z, z1.w,
                    z2.x, z2.y, z2.z, z2.w, z3.x, z3.y, z3.z, z3.w };
    float s = 0.f, s2 = 0.f;
#pragma unroll
    for (int c = 0; c < 16; ++c) { s += x[c]; s2 += x[c] * x[c]; }
    float m = s * (1.0f/16.0f);
    float r = rsqrtf(s2 * (1.0f/16.0f) - m*m + 1e-5f);
    float acc[4] = {0.f, 0.f, 0.f, 0.f};
#pragma unroll
    for (int c = 0; c < 16; ++c) {
        float y = (x[c] - m) * r * gz[c];
#pragma unroll
        for (int h = 0; h < 4; ++h) acc[h] += y * Wz[c * 4 + h];
    }
    int kk = idx & 127, q = (idx >> 7) & 31, nb = (idx >> 12) & 511, b = idx >> 21;
    long base = ((((long)(b * H_)) * NB_ + nb) * NK_ + kk) * NQ_ + q;  // [..,k,q]
#pragma unroll
    for (int h = 0; h < 4; ++h)
        bias_t[base + (long)h * NB_ * NQ_ * NK_] = acc[h];
}

// ---------------------------------------------------------------------------
// Kernel 6: local attention; 4 independent waves per 128-thread block.
// QK^T via WMMA (bias_t+mask preloaded into C via float4), wave softmax,
// P->LDS transpose, P@V via WMMA on transposed V, fused sigmoid gate.
// ---------------------------------------------------------------------------
__global__ void k_attn(const h16* __restrict__ Q, const h16* __restrict__ Km,
                       const h16* __restrict__ Vt, const float* __restrict__ g,
                       const float* __restrict__ bias_t, h16* __restrict__ go) {
    __shared__ __align__(16) h16 P[4][16 * NK_];
    int wv   = threadIdx.x >> 5;
    int gid  = blockIdx.x * 4 + wv;
    int qh   = gid & 1;
    int nb   = (gid >> 1) & 511;
    int h    = (gid >> 10) & 3;
    int b    = gid >> 12;
    int lane = threadIdx.x & 31;
    int half = lane >> 4, l15 = lane & 15;

    int  qrow0   = nb * NQ_ + qh * 16;
    long tokbase = (long)b * NTOK;
    int  kstart  = nb * NQ_ - 48;  // (NK-NQ)/2

    v16h qa = load_a_frag(Q, tokbase + qrow0 + l15, CA_, h * DH_, lane);
    long bb = (((long)(b * H_ + h)) * NB_ + nb) * (NQ_ * NK_);

    v8f sc[8];
#pragma unroll
    for (int t = 0; t < 8; ++t) {
        int key = t * 16 + l15;
        int gk  = kstart + key;
        bool valid = (gk >= 0) && (gk < NTOK);
        int tokc = gk < 0 ? 0 : (gk >= NTOK ? NTOK - 1 : gk);
        v8f c;
        if (valid) {
            const float4* bp =
                (const float4*)(bias_t + bb + (long)key * NQ_ + qh * 16 + 8 * half);
            float4 p0 = bp[0], p1 = bp[1];
            c[0] = p0.x; c[1] = p0.y; c[2] = p0.z; c[3] = p0.w;
            c[4] = p1.x; c[5] = p1.y; c[6] = p1.z; c[7] = p1.w;
        } else {
            c = (v8f)(-1.0e9f);
        }
        v16h kb = load_kb_frag(Km, tokbase + tokc, h * DH_, half);
        sc[t] = wmma_f16(qa, kb, c);
    }

    // row softmax over 128 keys (8 tiles x 16 lanes within half-wave)
    float inv[8];
#pragma unroll
    for (int r = 0; r < 8; ++r) {
        float mx = sc[0][r];
#pragma unroll
        for (int t = 1; t < 8; ++t) mx = fmaxf(mx, sc[t][r]);
#pragma unroll
        for (int m = 1; m < 16; m <<= 1) mx = fmaxf(mx, __shfl_xor(mx, m, 16));
        float sum = 0.f;
#pragma unroll
        for (int t = 0; t < 8; ++t) {
            float e = __expf(sc[t][r] - mx);
            sc[t][r] = e;
            sum += e;
        }
#pragma unroll
        for (int m = 1; m < 16; m <<= 1) sum += __shfl_xor(sum, m, 16);
        inv[r] = 1.0f / sum;
    }
#pragma unroll
    for (int t = 0; t < 8; ++t)
#pragma unroll
        for (int r = 0; r < 8; ++r)
            P[wv][(r + 8 * half) * NK_ + t * 16 + l15] = (h16)(sc[t][r] * inv[r]);
    __syncthreads();

    // O = P @ V (transposed V: contiguous-token b128 loads) ; fuse gate
#pragma unroll
    for (int dt = 0; dt < 2; ++dt) {
        int ch = h * DH_ + dt * 16 + l15;
        long chbase = (long)ch * MROWS + tokbase;
        v8f o = (v8f)(0.f);
#pragma unroll
        for (int kt = 0; kt < 4; ++kt) {
            v16h pa = load_a_frag(&P[wv][0], (long)l15, NK_, kt * 32, lane);
            v16h vb = load_vt_frag(Vt, chbase, kstart + kt * 32 + 16 * half, NTOK);
            o = wmma_f16(pa, vb, o);
        }
#pragma unroll
        for (int r = 0; r < 8; ++r) {
            long tok = tokbase + qrow0 + r + 8 * half;
            go[tok * CA_ + ch] = (h16)(g[tok * CA_ + ch] * o[r]);
        }
    }
}

// ---------------------------------------------------------------------------
// Kernel 7: plain GEMM, 4 row tiles per wave: C(f32) = A(f16,MxK) @ Wt(f16)
// ---------------------------------------------------------------------------
__global__ void k_gemm_f32out(const h16* __restrict__ A, const h16* __restrict__ Wt,
                              float* __restrict__ C, int K) {
    int wid  = blockIdx.x * (blockDim.x >> 5) + (threadIdx.x >> 5);
    int lane = threadIdx.x & 31;
    int half = lane >> 4, l15 = lane & 15;
    int tg = wid >> 3, tn = wid & 7;
    long row0 = (long)tg * 64 + l15;
    v8f c[4];
#pragma unroll
    for (int j = 0; j < 4; ++j) c[j] = (v8f)(0.f);
    for (int k = 0; k < K; k += 32) {
        v16h bf = load_bt_frag(Wt, k, tn * 16, K, lane);
#pragma unroll
        for (int j = 0; j < 4; ++j) {
            v16h af = load_a_frag(A, row0 + j * 16, K, k, lane);
            c[j] = wmma_f16(af, bf, c[j]);
        }
    }
    int n = tn * 16 + l15;
#pragma unroll
    for (int j = 0; j < 4; ++j)
#pragma unroll
        for (int r = 0; r < 8; ++r) {
            long orow = (long)tg * 64 + j * 16 + r + 8 * half;
            C[orow * CA_ + n] = c[j][r];
        }
}

// ---------------------------------------------------------------------------
// Kernel 8: SwiGLU dual GEMM, 2 row tiles per wave: t = silu(a3@Wa1)*(a3@Wa2)
// ---------------------------------------------------------------------------
__global__ void k_mlp_gemm(const h16* __restrict__ A, const h16* __restrict__ W1t,
                           const h16* __restrict__ W2t, h16* __restrict__ out) {
    int wid  = blockIdx.x * (blockDim.x >> 5) + (threadIdx.x >> 5);
    int lane = threadIdx.x & 31;
    int half = lane >> 4, l15 = lane & 15;
    int tg = wid >> 4, tn = wid & 15;  // 16 column tiles (N=256)
    long row0 = (long)tg * 32 + l15;
    v8f c1[2], c2[2];
#pragma unroll
    for (int j = 0; j < 2; ++j) { c1[j] = (v8f)(0.f); c2[j] = (v8f)(0.f); }
#pragma unroll
    for (int k = 0; k < CA_; k += 32) {
        v16h b1 = load_bt_frag(W1t, k, tn * 16, CA_, lane);
        v16h b2 = load_bt_frag(W2t, k, tn * 16, CA_, lane);
#pragma unroll
        for (int j = 0; j < 2; ++j) {
            v16h af = load_a_frag(A, row0 + j * 16, CA_, k, lane);
            c1[j] = wmma_f16(af, b1, c1[j]);
            c2[j] = wmma_f16(af, b2, c2[j]);
        }
    }
    int n = tn * 16 + l15;
#pragma unroll
    for (int j = 0; j < 2; ++j)
#pragma unroll
        for (int r = 0; r < 8; ++r) {
            long orow = (long)tg * 32 + j * 16 + r + 8 * half;
            float x = c1[j][r];
            out[orow * 256 + n] = (h16)((x * sigmoidf_(x)) * c2[j][r]);
        }
}

// ---------------------------------------------------------------------------
// Kernel 9: final combine dual GEMM from raw s, 4 row tiles per wave:
//   out = sigmoid(s@Wsl + bsl)*attn_out + sigmoid(s@Wsc + bsc)*u
// ---------------------------------------------------------------------------
__global__ void k_final_gemm(const h16* __restrict__ S, const h16* __restrict__ Wslt,
                             const h16* __restrict__ Wsct, const float* __restrict__ bsl,
                             const float* __restrict__ bsc, const float* __restrict__ ao,
                             const float* __restrict__ u, float* __restrict__ out) {
    int wid  = blockIdx.x * (blockDim.x >> 5) + (threadIdx.x >> 5);
    int lane = threadIdx.x & 31;
    int half = lane >> 4, l15 = lane & 15;
    int tg = wid >> 3, tn = wid & 7;
    long row0 = (long)tg * 64 + l15;
    v8f c1[4], c2[4];
#pragma unroll
    for (int j = 0; j < 4; ++j) { c1[j] = (v8f)(0.f); c2[j] = (v8f)(0.f); }
#pragma unroll
    for (int k = 0; k < CS_; k += 32) {
        v16h b1 = load_bt_frag(Wslt, k, tn * 16, CS_, lane);
        v16h b2 = load_bt_frag(Wsct, k, tn * 16, CS_, lane);
#pragma unroll
        for (int j = 0; j < 4; ++j) {
            v16h af = load_a_frag(S, row0 + j * 16, CS_, k, lane);
            c1[j] = wmma_f16(af, b1, c1[j]);
            c2[j] = wmma_f16(af, b2, c2[j]);
        }
    }
    int n = tn * 16 + l15;
#pragma unroll
    for (int j = 0; j < 4; ++j)
#pragma unroll
        for (int r = 0; r < 8; ++r) {
            long orow = (long)tg * 64 + j * 16 + r + 8 * half;
            out[orow * CA_ + n] = sigmoidf_(c1[j][r] + bsl[n]) * ao[orow * CA_ + n]
                                + sigmoidf_(c2[j][r] + bsc[n]) * u[orow * CA_ + n];
        }
}

// ---------------------------------------------------------------------------
// Host launch
// ---------------------------------------------------------------------------
extern "C" void kernel_launch(void* const* d_in, const int* in_sizes, int n_in,
                              void* d_out, int out_size, void* d_ws, size_t ws_size,
                              hipStream_t stream) {
    const float* a        = (const float*)d_in[0];
    const float* s        = (const float*)d_in[1];
    const float* z        = (const float*)d_in[2];
    const float* apb_gs   = (const float*)d_in[3];
    const float* apb_Wg   = (const float*)d_in[4];
    const float* apb_bg   = (const float*)d_in[5];
    const float* apb_Wsk  = (const float*)d_in[6];
    const float* Wq       = (const float*)d_in[7];
    const float* bq       = (const float*)d_in[8];
    const float* Wk       = (const float*)d_in[9];
    const float* Wv       = (const float*)d_in[10];
    const float* Wgate    = (const float*)d_in[11];
    const float* Wo       = (const float*)d_in[12];
    const float* gamma_z  = (const float*)d_in[13];
    const float* Wz       = (const float*)d_in[14];
    const float* Ws_last  = (const float*)d_in[15];
    const float* bs_last  = (const float*)d_in[16];
    const float* ct_gs    = (const float*)d_in[17];
    const float* ct_Wg    = (const float*)d_in[18];
    const float* ct_bg    = (const float*)d_in[19];
    const float* ct_Wsk   = (const float*)d_in[20];
    const float* Wa1      = (const float*)d_in[21];
    const float* Wa2      = (const float*)d_in[22];
    const float* Wb       = (const float*)d_in[23];
    const float* Ws       = (const float*)d_in[24];
    const float* bs       = (const float*)d_in[25];
    float* out = (float*)d_out;

    char* ws = (char*)d_ws;
    size_t off = 0;
    auto alloc = [&](size_t bytes) { size_t o = off; off = (off + bytes + 255) & ~(size_t)255; return o; };

    const size_t MC = (size_t)MROWS * CA_;
    size_t o_aln  = alloc(MC * 4);        // LN(a) f32
    size_t o_sna  = alloc(MC * 2);        // LN(s)*apb_gamma f16
    size_t o_snc  = alloc(MC * 2);        // LN(s)*ct_gamma  f16
    size_t o_sh   = alloc(MC * 2);        // raw s f16
    size_t o_a1   = alloc(MC * 2);
    size_t o_a3   = alloc(MC * 2);
    size_t o_q    = alloc(MC * 2);
    size_t o_k    = alloc(MC * 2);
    size_t o_vt   = alloc(MC * 2);        // V transposed [ch][tok]
    size_t o_g    = alloc(MC * 4);        // sigmoid gate f32
    size_t o_bias = alloc((size_t)2 * H_ * NB_ * NQ_ * NK_ * 4);  // 67 MB, [.,k,q]
    size_t o_go   = alloc(MC * 2);        // g*o f16
    size_t o_ao   = alloc(MC * 4);        // attn_out f32
    size_t o_t    = alloc((size_t)MROWS * 256 * 2);               // SwiGLU t f16
    size_t o_u    = alloc(MC * 4);
    // transposed f16 weights
    size_t w_apbg = alloc(128*128*2), w_apbs = alloc(128*128*2);
    size_t w_ctg  = alloc(128*128*2), w_cts  = alloc(128*128*2);
    size_t w_q    = alloc(128*128*2), w_k    = alloc(128*128*2);
    size_t w_v    = alloc(128*128*2), w_gate = alloc(128*128*2);
    size_t w_o    = alloc(128*128*2), w_sl   = alloc(128*128*2);
    size_t w_s    = alloc(128*128*2);
    size_t w_a1   = alloc(128*256*2), w_a2   = alloc(128*256*2);
    size_t w_b    = alloc(256*128*2);

    auto convT = [&](const float* src, size_t dst, int K, int N) {
        int n = K * N;
        k_f32_to_f16_t<<<(n + 255) / 256, 256, 0, stream>>>(src, (h16*)(ws + dst), K, N);
    };
    convT(apb_Wg, w_apbg, 128, 128);  convT(apb_Wsk, w_apbs, 128, 128);
    convT(ct_Wg,  w_ctg,  128, 128);  convT(ct_Wsk,  w_cts,  128, 128);
    convT(Wq, w_q, 128, 128); convT(Wk, w_k, 128, 128); convT(Wv, w_v, 128, 128);
    convT(Wgate, w_gate, 128, 128); convT(Wo, w_o, 128, 128);
    convT(Ws_last, w_sl, 128, 128); convT(Ws, w_s, 128, 128);
    convT(Wa1, w_a1, 128, 256); convT(Wa2, w_a2, 128, 256);
    convT(Wb, w_b, 256, 128);

    // 1) row LayerNorms
    k_rowln<<<MROWS / 8, 256, 0, stream>>>(a, s, apb_gs, ct_gs,
        (float*)(ws + o_aln), (h16*)(ws + o_sna), (h16*)(ws + o_snc), (h16*)(ws + o_sh));

    // 2) a1 = adaLN(apb):  512 row groups x 8 col tiles = 4096 waves
    k_adaln_gemm<<<1024, 128, 0, stream>>>((const h16*)(ws + o_sna),
        (const h16*)(ws + w_apbg), (const h16*)(ws + w_apbs), apb_bg,
        (const float*)(ws + o_aln), (h16*)(ws + o_a1));

    // 3) q,k,vt,g from a1: 1024 row groups x 8 = 8192 waves
    k_qkvg_gemm<<<2048, 128, 0, stream>>>((const h16*)(ws + o_a1),
        (const h16*)(ws + w_q), (const h16*)(ws + w_k),
        (const h16*)(ws + w_v), (const h16*)(ws + w_gate), bq,
        (h16*)(ws + o_q), (h16*)(ws + o_k), (h16*)(ws + o_vt), (float*)(ws + o_g));

    // 4) pair bias from z (HBM-bound stream)
    k_bias<<<4194304 / 256, 256, 0, stream>>>(z, gamma_z, Wz, (float*)(ws + o_bias));

    // 5) local attention + fused gate (8192 waves, 4 per block)
    k_attn<<<2048, 128, 0, stream>>>((const h16*)(ws + o_q), (const h16*)(ws + o_k),
        (const h16*)(ws + o_vt), (const float*)(ws + o_g),
        (const float*)(ws + o_bias), (h16*)(ws + o_go));

    // 6) attn_out = (g*o) @ Wo
    k_gemm_f32out<<<1024, 128, 0, stream>>>((const h16*)(ws + o_go),
        (const h16*)(ws + w_o), (float*)(ws + o_ao), 128);

    // 7) a3 = adaLN(ct)
    k_adaln_gemm<<<1024, 128, 0, stream>>>((const h16*)(ws + o_snc),
        (const h16*)(ws + w_ctg), (const h16*)(ws + w_cts), ct_bg,
        (const float*)(ws + o_aln), (h16*)(ws + o_a3));

    // 8) t = silu(a3@Wa1) * (a3@Wa2): 1024 x 16 = 16384 waves
    k_mlp_gemm<<<4096, 128, 0, stream>>>((const h16*)(ws + o_a3),
        (const h16*)(ws + w_a1), (const h16*)(ws + w_a2), (h16*)(ws + o_t));

    // 9) u = t @ Wb (K=256)
    k_gemm_f32out<<<1024, 128, 0, stream>>>((const h16*)(ws + o_t),
        (const h16*)(ws + w_b), (float*)(ws + o_u), 256);

    // 10) out = sigmoid(s@Ws_last+bsl)*attn_out + sigmoid(s@Ws+bs)*u
    k_final_gemm<<<1024, 128, 0, stream>>>((const h16*)(ws + o_sh),
        (const h16*)(ws + w_sl), (const h16*)(ws + w_s), bs_last, bs,
        (const float*)(ws + o_ao), (const float*)(ws + o_u), out);
}
